// AtomicaDynamics_6493990552278
// MI455X (gfx1250) — matrix operand order repaired and checked by reading, not compiled, not run on hardware
//
#include <hip/hip_runtime.h>
#include <hip/hip_bf16.h>

// ======================= CDNA5 / gfx1250 EGNN dynamics ======================
// Strategy: everything fits in L2 (192MB) -> compute-bound (~80 GFLOP of small
// GEMMs). Use v_wmma_f32_16x16x32_f16 for all MLP GEMMs. Weights are packed
// once per launch into the ISA B-fragment layout (f16); activations staged to
// LDS as f16 tiles and read back as A-fragments per the ISA 16x32 A layout.
// NOTE: the reference's 1e-4 jitter uses JAX PRNG and is omitted here.

typedef _Float16 h16;
typedef _Float16 v16h __attribute__((ext_vector_type(16)));
typedef float    v8f  __attribute__((ext_vector_type(8)));

#define N_LIG 1024
#define N_POC 3072
#define E_LL  64512
#define E_LP  196608

__device__ __forceinline__ float silu_f(float x){ return x / (1.f + __expf(-x)); }

__device__ __forceinline__ v8f wmma_f16(v16h a, v16h b, v8f c){
  // (neg_a, A, neg_b, B, c_mod, C, reuse_a, reuse_b)
  return __builtin_amdgcn_wmma_f32_16x16x32_f16(false, a, false, b, (short)0, c, false, false);
}

// A-matrix fragment (16x32 f16, M=rows, K=cols) from LDS tile, row stride rs.
__device__ __forceinline__ v16h load_a_frag(const h16* Alds, int rs, int kt, int lane){
  int l = lane & 15, hi = lane >> 4;
  const h16* base = Alds + l*rs + kt*32;
  v16h a;
  #pragma unroll
  for (int v = 0; v < 8; ++v){
    int k0 = ((v>>2)<<4) + (hi<<3) + ((v&3)<<1);
    a[2*v]   = base[k0];
    a[2*v+1] = base[k0+1];
  }
  return a;
}

// B-matrix fragment: pre-packed so each lane reads 16 contiguous halfs (32B).
__device__ __forceinline__ v16h load_b_frag(const h16* pk, int frag, int lane){
  const v16h* p = reinterpret_cast<const v16h*>(pk + (size_t)frag*512);
  return p[lane];
}

// Pack f32 weight W[din][dout] (row-major, K x N) -> f16 fragments.
// Fragment f = kt*(Np/16)+nt; within fragment, half index (lane,h) holds
// B[kt*32 + (h>>3)*16 + (lane>>4)*8 + (h&7)][nt*16 + (lane&15)]  (zero-padded).
__global__ void pack_kernel(const float* __restrict__ W, int din, int dout,
                            int Kp, int Np, h16* __restrict__ dst){
  int idx = blockIdx.x*blockDim.x + threadIdx.x;
  int total = (Kp>>5)*(Np>>4)*512;
  if (idx >= total) return;
  int f = idx >> 9, w = idx & 511;
  int lane = w >> 4, hh = w & 15;
  int nts = Np >> 4;
  int kt = f / nts, nt = f - kt*nts;
  int l = lane & 15, hi = lane >> 4;
  int K = kt*32 + ((hh>>3)<<4) + (hi<<3) + (hh&7);
  int N = nt*16 + l;
  float v = (K < din && N < dout) ? W[(size_t)K*dout + N] : 0.f;
  dst[idx] = (h16)v;
}

// Generic Y[M][dout] = act(X[M][din] @ Wpk + bias) (+ res). 4 waves/block.
__global__ __launch_bounds__(128) void gemm_kernel(
    const float* __restrict__ X, int xs, int M,
    const h16* __restrict__ Wpk, const float* __restrict__ bias,
    int din, int Kp, int dout, int Np,
    float* __restrict__ Y, int ys, int act,
    const float* __restrict__ res, int rs)
{
  __shared__ h16 A[4*16*256];
  int lane = threadIdx.x & 31, w = threadIdx.x >> 5;
  int tile = blockIdx.x*4 + w;
  if (tile*16 >= M) return;
  int l = lane & 15, hi = lane >> 4;
  h16* Aw = A + w*4096;
  const float* src = X + (size_t)(tile*16 + l)*xs;
  int half = Kp >> 1;
  for (int j = hi*half; j < hi*half + half; ++j)
    Aw[l*Kp + j] = (j < din) ? (h16)src[j] : (h16)0.f;
  int nts = Np >> 4, kts = Kp >> 5;
  for (int nt = 0; nt < nts; ++nt){
    int N = nt*16 + l;
    float bv = (bias && N < dout) ? bias[N] : 0.f;
    v8f acc;
    #pragma unroll
    for (int q = 0; q < 8; ++q) acc[q] = bv;
    for (int kt = 0; kt < kts; ++kt)
      acc = wmma_f16(load_a_frag(Aw, Kp, kt, lane), load_b_frag(Wpk, kt*nts + nt, lane), acc);
    if (N < dout){
      #pragma unroll
      for (int q = 0; q < 8; ++q){
        int Mr = tile*16 + (hi ? q + 8 : q);
        float x = acc[q];
        if (act) x = silu_f(x);
        if (res) x += res[(size_t)Mr*rs + N];
        Y[(size_t)Mr*ys + N] = x;
      }
    }
  }
}

// Fused edge MLP. 1 wave = 16 edges. Input per edge:
// [ hq[row](64) | hk[col](64) | radial,d0,emb8 (10) | pad -> 160 ].
// Layer1 160->64 SiLU, Layer2 64->64 SiLU.
// EQ=false: atomic scatter m/100 into agg[row]. EQ=true: s=tanh(z.wout)*3.75,
// scatter cdiff*s/100 into xout[row].
template<bool EQ>
__global__ __launch_bounds__(256) void edge_kernel(
    const float* __restrict__ hq, int hqs,
    const float* __restrict__ hk, int hks,
    const float* __restrict__ xq, const float* __restrict__ xk,
    const int* __restrict__ rows, const int* __restrict__ cols, int E,
    const float* __restrict__ d0, const float* __restrict__ emb8,
    const h16* __restrict__ w0pk, const float* __restrict__ b0,
    const h16* __restrict__ w1pk, const float* __restrict__ b1,
    const float* __restrict__ wout,
    float* __restrict__ agg, int aggs,
    float* __restrict__ xout)
{
  __shared__ h16  A1[8*16*160];
  __shared__ h16  A2[8*16*64];
  __shared__ float CD[8][16][3];
  __shared__ float SS[8][16];
  int lane = threadIdx.x & 31, w = threadIdx.x >> 5;
  int tile = blockIdx.x*8 + w;
  int e0i = tile*16;
  if (e0i >= E) return;
  int l = lane & 15, hi = lane >> 4;
  int e = e0i + l;
  int r = rows[e], c = cols[e];
  h16* A = A1 + w*2560;
  h16* Z = A2 + w*1024;
  if (hi == 0){
    const float* s = hq + (size_t)r*hqs;
    #pragma unroll 8
    for (int j = 0; j < 64; ++j) A[l*160 + j] = (h16)s[j];
  } else {
    const float* s = hk + (size_t)c*hks;
    #pragma unroll 8
    for (int j = 0; j < 64; ++j) A[l*160 + 64 + j] = (h16)s[j];
    float dx = xq[r*3+0] - xk[c*3+0];
    float dy = xq[r*3+1] - xk[c*3+1];
    float dz = xq[r*3+2] - xk[c*3+2];
    float rad = dx*dx + dy*dy + dz*dz;
    A[l*160+128] = (h16)rad;
    A[l*160+129] = (h16)d0[e];
    #pragma unroll
    for (int j = 0; j < 8; ++j) A[l*160+130+j] = (h16)emb8[j];
    #pragma unroll
    for (int j = 138; j < 160; ++j) A[l*160+j] = (h16)0.f;
    if (EQ){
      float inv = 1.f/(sqrtf(rad + 1e-8f) + 1.f);   // NORM_CONST = 1.0
      CD[w][l][0] = dx*inv; CD[w][l][1] = dy*inv; CD[w][l][2] = dz*inv;
    }
  }
  // same-wave LDS ops are in order; no cross-wave sharing -> no barrier.
  // ---- layer 1: 160 -> 64, SiLU -> Z (f16) ----
  for (int nt = 0; nt < 4; ++nt){
    float bv = b0[nt*16 + l];
    v8f acc;
    #pragma unroll
    for (int q = 0; q < 8; ++q) acc[q] = bv;
    #pragma unroll
    for (int kt = 0; kt < 5; ++kt)
      acc = wmma_f16(load_a_frag(A, 160, kt, lane), load_b_frag(w0pk, kt*4 + nt, lane), acc);
    #pragma unroll
    for (int q = 0; q < 8; ++q){
      int Mr = hi ? q + 8 : q;
      Z[Mr*64 + nt*16 + l] = (h16)silu_f(acc[q]);
    }
  }
  // ---- layer 2: 64 -> 64, SiLU ----
  float part[8];
  #pragma unroll
  for (int q = 0; q < 8; ++q) part[q] = 0.f;
  for (int nt = 0; nt < 4; ++nt){
    int N = nt*16 + l;
    float bv = b1[N];
    v8f acc;
    #pragma unroll
    for (int q = 0; q < 8; ++q) acc[q] = bv;
    #pragma unroll
    for (int kt = 0; kt < 2; ++kt)
      acc = wmma_f16(load_a_frag(Z, 64, kt, lane), load_b_frag(w1pk, kt*4 + nt, lane), acc);
    if (EQ){
      float wv = wout[N];
      #pragma unroll
      for (int q = 0; q < 8; ++q) part[q] += silu_f(acc[q]) * wv;
    } else {
      #pragma unroll
      for (int q = 0; q < 8; ++q){
        int Mr = hi ? q + 8 : q;
        atomicAdd(&agg[(size_t)rows[e0i + Mr]*aggs + N], silu_f(acc[q]) * 0.01f);
      }
    }
  }
  if (EQ){
    #pragma unroll
    for (int q = 0; q < 8; ++q){
      float p = part[q];
      p += __shfl_xor(p, 1, 32);
      p += __shfl_xor(p, 2, 32);
      p += __shfl_xor(p, 4, 32);
      p += __shfl_xor(p, 8, 32);
      part[q] = p;
    }
    if (l == 0){
      #pragma unroll
      for (int q = 0; q < 8; ++q) SS[w][hi*8 + q] = part[q];
    }
    for (int k = lane; k < 48; k += 32){
      int ei = k/3, d = k - 3*ei;
      float s = tanhf(SS[w][ei]) * 3.75f * 0.01f;   // COORDS_RANGE/NORM_FACTOR
      atomicAdd(&xout[(size_t)rows[e0i + ei]*3 + d], CD[w][ei][d]*s);
    }
  }
}

// ---------------- small utility kernels ----------------
__global__ void ln_kernel(const float* __restrict__ X, int xs, int F,
                          float* __restrict__ Y, int N){
  int i = blockIdx.x*blockDim.x + threadIdx.x;
  if (i >= N) return;
  const float* x = X + (size_t)i*xs;
  float m = 0.f;
  for (int j = 0; j < F; ++j) m += x[j];
  m /= F;
  float v = 0.f;
  for (int j = 0; j < F; ++j){ float d = x[j]-m; v += d*d; }
  v /= F;
  float inv = rsqrtf(v + 1e-5f);
  float* y = Y + (size_t)i*F;
  for (int j = 0; j < F; ++j) y[j] = (x[j]-m)*inv;
}
__global__ void setcol_kernel(float* Y, int ys, int col, const float* t, int N){
  int i = blockIdx.x*blockDim.x + threadIdx.x;
  if (i < N) Y[(size_t)i*ys + col] = t[0];
}
__global__ void copy3_kernel(const float* X, int xs, float* Y, int N3){
  int k = blockIdx.x*blockDim.x + threadIdx.x;
  if (k >= N3) return;
  int i = k/3, d = k - 3*i;
  Y[k] = X[(size_t)i*xs + d];
}
__global__ void zerocols_kernel(float* Y, int ys, int c0, int nc, int Ntot){
  int k = blockIdx.x*blockDim.x + threadIdx.x;
  if (k >= Ntot) return;
  int i = k/nc, j = k - i*nc;
  Y[(size_t)i*ys + c0 + j] = 0.f;
}
__global__ void d0_kernel(const float* xq, const float* xk,
                          const int* rows, const int* cols, int E, float* d0){
  int e = blockIdx.x*blockDim.x + threadIdx.x;
  if (e >= E) return;
  int r = rows[e], c = cols[e];
  float dx = xq[r*3]-xk[c*3], dy = xq[r*3+1]-xk[c*3+1], dz = xq[r*3+2]-xk[c*3+2];
  d0[e] = dx*dx + dy*dy + dz*dz;
}
__global__ void vel_kernel(const float* xfin, const float* xh, float* out, int N3){
  int k = blockIdx.x*blockDim.x + threadIdx.x;
  if (k >= N3) return;
  int i = k/3, d = k - 3*i;
  out[(size_t)i*67 + d] = xfin[k] - xh[(size_t)i*67 + d];
}

// ============================== host driver ==============================
extern "C" void kernel_launch(void* const* d_in, const int* in_sizes, int n_in,
                              void* d_out, int out_size, void* d_ws, size_t ws_size,
                              hipStream_t stream)
{
  (void)in_sizes; (void)n_in; (void)out_size; (void)ws_size;
  const float* xh_lig = (const float*)d_in[0];   // [1024][67]
  const float* xh_ctx = (const float*)d_in[1];   // [3072][131]
  const float* tptr   = (const float*)d_in[2];   // [1]
  const int*   ell    = (const int*)d_in[5];     // [2][64512]
  const int*   elp    = (const int*)d_in[6];     // [2][196608]
  const float* P      = (const float*)d_in[7];   // flattened params
  const int *rows_ll = ell, *cols_ll = ell + E_LL;
  const int *rows_lp = elp, *cols_lp = elp + E_LP;

  // ---- parameter offsets: jax pytree flatten (dict keys sorted; W then b;
  // bias=None dropped). Top: atom_dec, atom_enc, cross, ctx_enc, edge_emb, egnn.
  size_t off = 0;
  auto take = [&](size_t n){ size_t o = off; off += n; return o; };
  struct BO { size_t m0W,m0b,m1W,m1b,outW,e0W[2],e0b[2],e1W[2],e1b[2],n0W[2],n0b[2],n1W[2],n1b[2]; };
  auto takeBlk = [&](BO& b){
    b.m0W=take(138*64); b.m0b=take(64); b.m1W=take(64*64); b.m1b=take(64); b.outW=take(64);
    for (int g = 0; g < 2; ++g){
      b.e0W[g]=take(138*64); b.e0b[g]=take(64); b.e1W[g]=take(64*64); b.e1b[g]=take(64);
      b.n0W[g]=take(128*64); b.n0b[g]=take(64); b.n1W[g]=take(64*64); b.n1b[g]=take(64);
    }
  };
  size_t dec0W=take(64*128), dec0b=take(128), dec1W=take(128*64), dec1b=take(64);
  size_t enc0W=take(64*128), enc0b=take(128), enc1W=take(128*64), enc1b=take(64);
  BO cb[4]; for (int i = 0; i < 4; ++i) takeBlk(cb[i]);
  size_t kvW=take(65*64), kvb=take(64);
  size_t coW=take(64*65), cob=take(65);
  size_t qW =take(65*64), qb =take(64);
  size_t ctx0W=take(128*256), ctx0b=take(256), ctx1W=take(256*64), ctx1b=take(64);
  size_t eembO=take(16);
  BO eb[4]; for (int i = 0; i < 4; ++i) takeBlk(eb[i]);
  size_t embW=take(65*64), embb=take(64);
  size_t eoW =take(64*65), eob =take(65);

  // ---- workspace layout ----
  size_t cur = 0;
  auto walloc = [&](size_t bytes)->void*{
    void* p = (char*)d_ws + cur; cur += (bytes + 255) & ~(size_t)255; return p;
  };
  h16* PK = (h16*)walloc(2*1024*1024);
  h16* pkcur = PK;
  auto pack = [&](size_t woff, int din, int dout, int Kp, int Np)->const h16*{
    h16* dst = pkcur;
    int total = (Kp/32)*(Np/16)*512;
    pkcur += total;
    pack_kernel<<<(total+255)/256, 256, 0, stream>>>(P + woff, din, dout, Kp, Np, dst);
    return dst;
  };
  const h16 *pk_enc0 = pack(enc0W, 64,128, 64,128), *pk_enc1 = pack(enc1W,128, 64,128, 64);
  const h16 *pk_ctx0 = pack(ctx0W,128,256,128,256), *pk_ctx1 = pack(ctx1W,256, 64,256, 64);
  const h16 *pk_emb  = pack(embW,  65, 64, 96, 64), *pk_eo   = pack(eoW,  64, 65, 64, 80);
  const h16 *pk_q    = pack(qW,    65, 64, 96, 64), *pk_kv   = pack(kvW,  65, 64, 96, 64);
  const h16 *pk_co   = pack(coW,   64, 65, 64, 80);
  const h16 *pk_dec0 = pack(dec0W, 64,128, 64,128), *pk_dec1 = pack(dec1W,128, 64,128, 64);
  struct BP { const h16 *m0,*m1,*e0[2],*e1[2],*n0[2],*n1[2]; };
  BP ebp[4], cbp[4];
  auto packBlk = [&](const BO& b, BP& p){
    p.m0 = pack(b.m0W,138,64,160,64); p.m1 = pack(b.m1W,64,64,64,64);
    for (int g = 0; g < 2; ++g){
      p.e0[g] = pack(b.e0W[g],138,64,160,64); p.e1[g] = pack(b.e1W[g],64,64,64,64);
      p.n0[g] = pack(b.n0W[g],128,64,128,64); p.n1[g] = pack(b.n1W[g],64,64,64,64);
    }
  };
  for (int i = 0; i < 4; ++i){ packBlk(eb[i], ebp[i]); packBlk(cb[i], cbp[i]); }

  float* lnl   = (float*)walloc((size_t)N_LIG*64*4);
  float* lnp   = (float*)walloc((size_t)N_POC*128*4);
  float* mid   = (float*)walloc((size_t)N_POC*256*4);
  float* hl65  = (float*)walloc((size_t)N_LIG*65*4);
  float* hp65  = (float*)walloc((size_t)N_POC*65*4);
  float* H     = (float*)walloc((size_t)N_LIG*128*4);   // cols 0..63 h, 64..127 agg
  float* HK    = (float*)walloc((size_t)N_POC*64*4);
  float* hll65 = (float*)walloc((size_t)N_LIG*65*4);
  float* hlp65 = (float*)walloc((size_t)N_LIG*65*4);
  float* XA    = (float*)walloc((size_t)N_LIG*3*4);
  float* XB    = (float*)walloc((size_t)N_LIG*3*4);
  float* XP    = (float*)walloc((size_t)N_POC*3*4);
  float* D0L   = (float*)walloc((size_t)E_LL*4);
  float* D0P   = (float*)walloc((size_t)E_LP*4);

  auto gemm = [&](const float* X, int xs, int M, const h16* Wpk, const float* bias,
                  int din, int Kp, int dout, int Np, float* Y, int ys, int act,
                  const float* res, int rs){
    gemm_kernel<<<(M/16 + 3)/4, 128, 0, stream>>>(X, xs, M, Wpk, bias, din, Kp, dout, Np, Y, ys, act, res, rs);
  };

  // ---- encoders ----
  ln_kernel<<<(N_LIG+255)/256,256,0,stream>>>(xh_lig+3, 67, 64, lnl, N_LIG);
  gemm(lnl, 64, N_LIG, pk_enc0, P+enc0b, 64, 64, 128, 128, mid, 128, 1, nullptr, 0);
  gemm(mid,128, N_LIG, pk_enc1, P+enc1b,128,128,  64,  64, hl65, 65, 0, nullptr, 0);
  setcol_kernel<<<(N_LIG+255)/256,256,0,stream>>>(hl65, 65, 64, tptr, N_LIG);
  ln_kernel<<<(N_POC+255)/256,256,0,stream>>>(xh_ctx+3, 131, 128, lnp, N_POC);
  gemm(lnp,128, N_POC, pk_ctx0, P+ctx0b,128,128, 256, 256, mid, 256, 1, nullptr, 0);
  gemm(mid,256, N_POC, pk_ctx1, P+ctx1b,256,256,  64,  64, hp65, 65, 0, nullptr, 0);
  setcol_kernel<<<(N_POC+255)/256,256,0,stream>>>(hp65, 65, 64, tptr, N_POC);

  // ---- EGNN (ligand-ligand) ----
  copy3_kernel<<<(N_LIG*3+255)/256,256,0,stream>>>(xh_lig, 67, XA, N_LIG*3);
  gemm(hl65, 65, N_LIG, pk_emb, P+embb, 65, 96, 64, 64, H, 128, 0, nullptr, 0);
  d0_kernel<<<(E_LL+255)/256,256,0,stream>>>(XA, XA, rows_ll, cols_ll, E_LL, D0L);
  const float* emb_ll = P + eembO + 8;   // edge_emb row 1
  for (int L = 0; L < 4; ++L){
    for (int g = 0; g < 2; ++g){
      zerocols_kernel<<<(N_LIG*64+255)/256,256,0,stream>>>(H, 128, 64, 64, N_LIG*64);
      edge_kernel<false><<<E_LL/128, 256, 0, stream>>>(H,128, H,128, XA, XA,
          rows_ll, cols_ll, E_LL, D0L, emb_ll,
          ebp[L].e0[g], P+eb[L].e0b[g], ebp[L].e1[g], P+eb[L].e1b[g],
          nullptr, H+64, 128, nullptr);
      gemm(H, 128, N_LIG, ebp[L].n0[g], P+eb[L].n0b[g], 128,128,64,64, mid, 64, 1, nullptr, 0);
      gemm(mid, 64, N_LIG, ebp[L].n1[g], P+eb[L].n1b[g],  64, 64,64,64, H, 128, 0, H, 128);
    }
    hipMemcpyAsync(XB, XA, (size_t)N_LIG*3*4, hipMemcpyDeviceToDevice, stream);
    edge_kernel<true><<<E_LL/128, 256, 0, stream>>>(H,128, H,128, XA, XA,
        rows_ll, cols_ll, E_LL, D0L, emb_ll,
        ebp[L].m0, P+eb[L].m0b, ebp[L].m1, P+eb[L].m1b,
        P+eb[L].outW, nullptr, 0, XB);
    float* t = XA; XA = XB; XB = t;
  }
  gemm(H, 128, N_LIG, pk_eo, P+eob, 64, 64, 65, 80, hll65, 65, 0, nullptr, 0);

  // ---- cross EGNN (ligand <- pocket) ----
  gemm(hll65, 65, N_LIG, pk_q,  P+qb,  65, 96, 64, 64, H, 128, 0, nullptr, 0);
  gemm(hp65,  65, N_POC, pk_kv, P+kvb, 65, 96, 64, 64, HK, 64, 0, nullptr, 0);
  copy3_kernel<<<(N_POC*3+255)/256,256,0,stream>>>(xh_ctx, 131, XP, N_POC*3);
  d0_kernel<<<(E_LP+255)/256,256,0,stream>>>(XA, XP, rows_lp, cols_lp, E_LP, D0P);
  const float* emb_lp = P + eembO;       // edge_emb row 0
  for (int L = 0; L < 4; ++L){
    for (int g = 0; g < 2; ++g){
      zerocols_kernel<<<(N_LIG*64+255)/256,256,0,stream>>>(H, 128, 64, 64, N_LIG*64);
      edge_kernel<false><<<E_LP/128, 256, 0, stream>>>(H,128, HK,64, XA, XP,
          rows_lp, cols_lp, E_LP, D0P, emb_lp,
          cbp[L].e0[g], P+cb[L].e0b[g], cbp[L].e1[g], P+cb[L].e1b[g],
          nullptr, H+64, 128, nullptr);
      gemm(H, 128, N_LIG, cbp[L].n0[g], P+cb[L].n0b[g], 128,128,64,64, mid, 64, 1, nullptr, 0);
      gemm(mid, 64, N_LIG, cbp[L].n1[g], P+cb[L].n1b[g],  64, 64,64,64, H, 128, 0, H, 128);
    }
    hipMemcpyAsync(XB, XA, (size_t)N_LIG*3*4, hipMemcpyDeviceToDevice, stream);
    edge_kernel<true><<<E_LP/128, 256, 0, stream>>>(H,128, HK,64, XA, XP,
        rows_lp, cols_lp, E_LP, D0P, emb_lp,
        cbp[L].m0, P+cb[L].m0b, cbp[L].m1, P+cb[L].m1b,
        P+cb[L].outW, nullptr, 0, XB);
    float* t = XA; XA = XB; XB = t;
  }
  gemm(H, 128, N_LIG, pk_co, P+cob, 64, 64, 65, 80, hlp65, 65, 0, nullptr, 0);

  // ---- decoder + outputs ----
  float* out = (float*)d_out;
  gemm(hlp65, 65, N_LIG, pk_dec0, P+dec0b, 64, 64, 128, 128, mid, 128, 1, nullptr, 0);
  gemm(mid,  128, N_LIG, pk_dec1, P+dec1b,128,128,  64,  64, out+3, 67, 0, nullptr, 0);
  vel_kernel<<<(N_LIG*3+255)/256,256,0,stream>>>(XA, xh_lig, out, N_LIG*3);
  hipMemsetAsync(out + (size_t)N_LIG*67, 0, (size_t)N_POC*131*4, stream);
}
